// PDC_Vnoc_GINConvNet_8108898255298
// MI455X (gfx1250) — compile-verified, matrix operand order checked by reading, not run on hardware
//
#include <hip/hip_runtime.h>
#include <hip/hip_bf16.h>

typedef __attribute__((ext_vector_type(16))) _Float16 v16h;
typedef __attribute__((ext_vector_type(8)))  _Float16 v8h;
typedef __attribute__((ext_vector_type(8)))  float    v8f;

#define N_NODES 16384
#define N_EDGES 65536
#define NB      256
#define SEQL    1000
#define PADW    1040       // padded position dim for y1/y2 (max read = 960+63+15 = 1038)
#define W1OUT   985
#define W2OUT   970
#define W3OUT   955
#define BN_EPS  1e-5f

// ---------------------------------------------------------------------------
// GIN graph layers (latency-bound, scalar path)
// ---------------------------------------------------------------------------
__global__ void k_gather(const float* __restrict__ h, const int* __restrict__ src,
                         const int* __restrict__ dst, float* __restrict__ agg, int IC) {
  int idx = blockIdx.x * blockDim.x + threadIdx.x;
  int total = N_EDGES * IC;
  if (idx >= total) return;
  int e = idx / IC, c = idx - e * IC;
  atomicAdd(&agg[(size_t)dst[e] * IC + c], h[(size_t)src[e] * IC + c]);
}

__global__ __launch_bounds__(128)
void k_gin_mlp(const float* __restrict__ in, const float* __restrict__ agg, int IC,
               const float* __restrict__ W1, const float* __restrict__ b1,
               const float* __restrict__ W2, const float* __restrict__ b2,
               const float* __restrict__ bng, const float* __restrict__ bnb,
               const float* __restrict__ bnm, const float* __restrict__ bnv,
               float* __restrict__ out) {
  __shared__ float sW1[78 * 32];
  __shared__ float sW2[32 * 32];
  __shared__ float sb1[32], sb2[32], sg[32], sbt[32], sm[32], sv[32];
  const int tid = threadIdx.x;
  for (int i = tid; i < IC * 32; i += 128) sW1[i] = W1[i];
  for (int i = tid; i < 32 * 32; i += 128) sW2[i] = W2[i];
  if (tid < 32) {
    sb1[tid] = b1[tid]; sb2[tid] = b2[tid];
    sg[tid] = bng[tid]; sbt[tid] = bnb[tid]; sm[tid] = bnm[tid]; sv[tid] = bnv[tid];
  }
  __syncthreads();
  const int n = blockIdx.x * 128 + tid;
  if (n >= N_NODES) return;
  float hid[32];
#pragma unroll
  for (int j = 0; j < 32; ++j) hid[j] = sb1[j];
  for (int c = 0; c < IC; ++c) {
    float t = in[(size_t)n * IC + c] + agg[(size_t)n * IC + c];
#pragma unroll
    for (int j = 0; j < 32; ++j) hid[j] += t * sW1[c * 32 + j];
  }
#pragma unroll
  for (int j = 0; j < 32; ++j) hid[j] = fmaxf(hid[j], 0.f);
  float o2[32];
#pragma unroll
  for (int j = 0; j < 32; ++j) o2[j] = sb2[j];
#pragma unroll
  for (int j = 0; j < 32; ++j) {
    float t = hid[j];
#pragma unroll
    for (int j2 = 0; j2 < 32; ++j2) o2[j2] += t * sW2[j * 32 + j2];
  }
#pragma unroll
  for (int j = 0; j < 32; ++j) {
    float v = fmaxf(o2[j], 0.f);
    v = (v - sm[j]) * (sg[j] * rsqrtf(sv[j] + BN_EPS)) + sbt[j];
    out[(size_t)n * 32 + j] = v;
  }
}

__global__ void k_pool(const float* __restrict__ h, const int* __restrict__ batch,
                       float* __restrict__ pooled) {
  int idx = blockIdx.x * blockDim.x + threadIdx.x;
  if (idx >= N_NODES * 32) return;
  int n = idx >> 5, c = idx & 31;
  atomicAdd(&pooled[(size_t)batch[n] * 32 + c], h[idx]);
}

// ---------------------------------------------------------------------------
// Generic small dense layer: out = act(bn(in @ W + b))
// ---------------------------------------------------------------------------
__global__ void k_dense(const float* __restrict__ in, const float* __restrict__ W,
                        const float* __restrict__ bias,
                        const float* __restrict__ bng, const float* __restrict__ bnb,
                        const float* __restrict__ bnm, const float* __restrict__ bnv,
                        float* __restrict__ out, int M, int K, int N, int relu) {
  int idx = blockIdx.x * blockDim.x + threadIdx.x;
  if (idx >= M * N) return;
  int m = idx / N, n = idx - m * N;
  const float* ir = in + (size_t)m * K;
  float s = 0.f;
  for (int k = 0; k < K; ++k) s += ir[k] * W[(size_t)k * N + n];
  s += bias[n];
  if (bng) s = (s - bnm[n]) * (bng[n] * rsqrtf(bnv[n] + BN_EPS)) + bnb[n];
  if (relu) s = fmaxf(s, 0.f);
  out[idx] = s;
}

__global__ void k_concat(const float* __restrict__ xd, const float* __restrict__ xcf,
                         float* __restrict__ z) {
  int idx = blockIdx.x * blockDim.x + threadIdx.x;
  if (idx >= NB * 256) return;
  int m = idx >> 8, n = idx & 255;
  z[idx] = (n < 128) ? xd[m * 128 + n] : xcf[m * 128 + (n - 128)];
}

// ---------------------------------------------------------------------------
// Conv weight pre-pack into per-lane WMMA B-fragment order
//   fragment slot s: v = s>>1, h = s&1 ; kl = (v>>2)*16 + (lane>>4)*8 + (v&3)*2 + h
// conv1: kk = c*16 + k (c-major) ; conv2/3: kk = k*32 + c (tap-major)
// ---------------------------------------------------------------------------
__global__ void k_pack_w1(const float* __restrict__ Wc, _Float16* __restrict__ pk) {
  int idx = blockIdx.x * blockDim.x + threadIdx.x;
  if (idx >= 2 * 128 * 32 * 16) return;
  int s = idx & 15, lane = (idx >> 4) & 31, ck = (idx >> 9) & 127, nt = idx >> 16;
  int v = s >> 1, h = s & 1;
  int kl = (v >> 2) * 16 + (lane >> 4) * 8 + (v & 3) * 2 + h;
  int kk = ck * 32 + kl;
  int c = kk >> 4, k = kk & 15;
  int o = (lane & 15) + nt * 16;
  pk[idx] = (_Float16)Wc[((size_t)o * 256 + c) * 16 + k];
}

__global__ void k_pack_w23(const float* __restrict__ Wc, _Float16* __restrict__ pk) {
  int idx = blockIdx.x * blockDim.x + threadIdx.x;
  if (idx >= 2 * 16 * 32 * 16) return;
  int s = idx & 15, lane = (idx >> 4) & 31, ck = (idx >> 9) & 15, nt = idx >> 13;
  int v = s >> 1, h = s & 1;
  int kl = (v >> 2) * 16 + (lane >> 4) * 8 + (v & 3) * 2 + h;
  int kk = ck * 32 + kl;
  int k = kk >> 5, c = kk & 31;
  int o = (lane & 15) + nt * 16;
  pk[idx] = (_Float16)Wc[((size_t)o * 32 + c) * 16 + k];
}

// ---------------------------------------------------------------------------
// Conv1: implicit GEMM, M=positions N=32 K=4096, fused embed-gather + concat
// one block = (b, 64-position tile); 4 waves, each owning a 16xM slice
// ---------------------------------------------------------------------------
__global__ __launch_bounds__(128)
void k_conv1(const int* __restrict__ target, const float* __restrict__ embed,
             const float* __restrict__ xd, const _Float16* __restrict__ w1pk,
             const float* __restrict__ bias, const float* __restrict__ bng,
             const float* __restrict__ bnb, const float* __restrict__ bnm,
             const float* __restrict__ bnv, _Float16* __restrict__ y1) {
  __shared__ _Float16 xwin[256][80];
  __shared__ int tgt[80];
  const int tid = threadIdx.x;
  const int b = blockIdx.y;
  const int p0 = blockIdx.x * 64;
  for (int i = tid; i < 80; i += 128) {
    int p = p0 + i;
    tgt[i] = (p < SEQL) ? target[(size_t)b * SEQL + p] : -1;
  }
  __syncthreads();
  for (int i = tid; i < 256 * 80; i += 128) {
    int c = i / 80, pp = i - c * 80;
    float v = 0.f;
    if (tgt[pp] >= 0)
      v = (c < 128) ? embed[(size_t)tgt[pp] * 128 + c] : xd[(size_t)b * 128 + (c - 128)];
    xwin[c][pp] = (_Float16)v;
  }
  __syncthreads();

  const int lane = tid & 31, wv = tid >> 5;
  const int mrow = lane & 15, halfL = lane >> 4;
  const int pwin = wv * 16 + mrow;
  v8f acc0 = {}, acc1 = {};
  for (int ck = 0; ck < 128; ++ck) {
    v16h a;
#pragma unroll
    for (int v = 0; v < 8; ++v) {
      int c = ck * 2 + (v >> 2);
      int k = halfL * 8 + (v & 3) * 2;
      a[2 * v]     = xwin[c][pwin + k];
      a[2 * v + 1] = xwin[c][pwin + k + 1];
    }
    v16h b0 = *(const v16h*)(w1pk + ((size_t)ck * 32 + lane) * 16);
    v16h b1 = *(const v16h*)(w1pk + ((size_t)(128 + ck) * 32 + lane) * 16);
    acc0 = __builtin_amdgcn_wmma_f32_16x16x32_f16(false, a, false, b0, (short)0, acc0, false, false);
    acc1 = __builtin_amdgcn_wmma_f32_16x16x32_f16(false, a, false, b1, (short)0, acc1, false, false);
  }
  const int n0 = lane & 15;
  const int mbase = (lane >> 4) * 8;
#pragma unroll
  for (int r = 0; r < 8; ++r) {
    int p = p0 + wv * 16 + mbase + r;
    if (p < W1OUT) {
      int o = n0;
      float v = acc0[r] + bias[o];
      v = (v - bnm[o]) * (bng[o] * rsqrtf(bnv[o] + BN_EPS)) + bnb[o];
      y1[((size_t)b * PADW + p) * 32 + o] = (_Float16)fmaxf(v, 0.f);
      o = n0 + 16;
      v = acc1[r] + bias[o];
      v = (v - bnm[o]) * (bng[o] * rsqrtf(bnv[o] + BN_EPS)) + bnb[o];
      y1[((size_t)b * PADW + p) * 32 + o] = (_Float16)fmaxf(v, 0.f);
    }
  }
}

// ---------------------------------------------------------------------------
// Conv2/Conv3: implicit GEMM K=512, A-fragments are two contiguous 16B loads
// store_y=1 -> write yout ; store_y=0 -> fused global max pool via atomicMax
// ---------------------------------------------------------------------------
__global__ __launch_bounds__(128)
void k_conv23(const _Float16* __restrict__ yin, const _Float16* __restrict__ wpk,
              const float* __restrict__ bias, const float* __restrict__ bng,
              const float* __restrict__ bnb, const float* __restrict__ bnm,
              const float* __restrict__ bnv, _Float16* __restrict__ yout,
              float* __restrict__ xcm, int wout, int store_y) {
  const int tid = threadIdx.x, lane = tid & 31, wv = tid >> 5;
  const int b = blockIdx.y;
  const int p0 = blockIdx.x * 64;
  const int mrow = lane & 15, halfL = lane >> 4;
  const int prow = wv * 16 + mrow;
  const _Float16* ybase = yin + (size_t)b * PADW * 32;
  v8f acc0 = {}, acc1 = {};
  for (int ck = 0; ck < 16; ++ck) {
    const _Float16* row = ybase + (size_t)(p0 + prow + ck) * 32 + halfL * 8;
    v8h lo = *(const v8h*)(row);
    v8h hi = *(const v8h*)(row + 16);
    v16h a = __builtin_shufflevector(lo, hi, 0, 1, 2, 3, 4, 5, 6, 7,
                                     8, 9, 10, 11, 12, 13, 14, 15);
    v16h b0 = *(const v16h*)(wpk + ((size_t)ck * 32 + lane) * 16);
    v16h b1 = *(const v16h*)(wpk + ((size_t)(16 + ck) * 32 + lane) * 16);
    acc0 = __builtin_amdgcn_wmma_f32_16x16x32_f16(false, a, false, b0, (short)0, acc0, false, false);
    acc1 = __builtin_amdgcn_wmma_f32_16x16x32_f16(false, a, false, b1, (short)0, acc1, false, false);
  }
  const int n0 = lane & 15;
  const int mbase = (lane >> 4) * 8;
#pragma unroll
  for (int r = 0; r < 8; ++r) {
    int p = p0 + wv * 16 + mbase + r;
    if (p < wout) {
      float v0 = acc0[r] + bias[n0];
      v0 = fmaxf((v0 - bnm[n0]) * (bng[n0] * rsqrtf(bnv[n0] + BN_EPS)) + bnb[n0], 0.f);
      int o1 = n0 + 16;
      float v1 = acc1[r] + bias[o1];
      v1 = fmaxf((v1 - bnm[o1]) * (bng[o1] * rsqrtf(bnv[o1] + BN_EPS)) + bnb[o1], 0.f);
      if (store_y) {
        yout[((size_t)b * PADW + p) * 32 + n0] = (_Float16)v0;
        yout[((size_t)b * PADW + p) * 32 + o1] = (_Float16)v1;
      } else {
        // values are >= 0, so uint bit-pattern ordering == float ordering
        atomicMax((unsigned int*)&xcm[(size_t)b * 32 + n0], __float_as_uint(v0));
        atomicMax((unsigned int*)&xcm[(size_t)b * 32 + o1], __float_as_uint(v1));
      }
    }
  }
}

// ---------------------------------------------------------------------------
extern "C" void kernel_launch(void* const* d_in, const int* in_sizes, int n_in,
                              void* d_out, int out_size, void* d_ws, size_t ws_size,
                              hipStream_t stream) {
  const float* x       = (const float*)d_in[0];
  const int*   edge    = (const int*)d_in[1];
  const int*   batch   = (const int*)d_in[2];
  const int*   target  = (const int*)d_in[3];
  const float* g1_W1   = (const float*)d_in[4];
  const float* g1_b1   = (const float*)d_in[5];
  const float* g1_W2   = (const float*)d_in[6];
  const float* g1_b2   = (const float*)d_in[7];
  const float* gW1     = (const float*)d_in[8];
  const float* gb1     = (const float*)d_in[9];
  const float* gW2     = (const float*)d_in[10];
  const float* gb2     = (const float*)d_in[11];
  const float* bn_g    = (const float*)d_in[12];
  const float* bn_b    = (const float*)d_in[13];
  const float* bn_m    = (const float*)d_in[14];
  const float* bn_v    = (const float*)d_in[15];
  const float* fcxd_W  = (const float*)d_in[16];
  const float* fcxd_b  = (const float*)d_in[17];
  const float* embed   = (const float*)d_in[18];
  const float* c1_W    = (const float*)d_in[19];
  const float* c1_b    = (const float*)d_in[20];
  const float* c2_W    = (const float*)d_in[21];
  const float* c2_b    = (const float*)d_in[22];
  const float* c3_W    = (const float*)d_in[23];
  const float* c3_b    = (const float*)d_in[24];
  const float* cbn_g   = (const float*)d_in[25];
  const float* cbn_b   = (const float*)d_in[26];
  const float* cbn_m   = (const float*)d_in[27];
  const float* cbn_v   = (const float*)d_in[28];
  const float* fcxc_W  = (const float*)d_in[29];
  const float* fcxc_b  = (const float*)d_in[30];
  const float* bnfc_g  = (const float*)d_in[31];
  const float* bnfc_b  = (const float*)d_in[32];
  const float* bnfc_m  = (const float*)d_in[33];
  const float* bnfc_v  = (const float*)d_in[34];
  const float* fc1_W   = (const float*)d_in[35];
  const float* fc1_b   = (const float*)d_in[36];
  const float* fc2_W   = (const float*)d_in[37];
  const float* fc2_b   = (const float*)d_in[38];
  const float* out_W   = (const float*)d_in[39];
  const float* out_b   = (const float*)d_in[40];

  // workspace layout (256B aligned)
  char* ws = (char*)d_ws;
  size_t off = 0;
  auto alloc = [&](size_t bytes) { void* p = ws + off; off += (bytes + 255) & ~(size_t)255; return p; };
  float*    hA     = (float*)alloc((size_t)N_NODES * 32 * 4);
  float*    hB     = (float*)alloc((size_t)N_NODES * 32 * 4);
  float*    agg    = (float*)alloc((size_t)N_NODES * 78 * 4);
  float*    pooled = (float*)alloc((size_t)NB * 32 * 4);
  float*    xd     = (float*)alloc((size_t)NB * 128 * 4);
  float*    xcm    = (float*)alloc((size_t)NB * 32 * 4);
  float*    xcf    = (float*)alloc((size_t)NB * 128 * 4);
  float*    z      = (float*)alloc((size_t)NB * 256 * 4);
  float*    t1     = (float*)alloc((size_t)NB * 1024 * 4);
  float*    t2     = (float*)alloc((size_t)NB * 256 * 4);
  _Float16* w1pk   = (_Float16*)alloc((size_t)2 * 128 * 32 * 16 * 2);
  _Float16* w2pk   = (_Float16*)alloc((size_t)2 * 16 * 32 * 16 * 2);
  _Float16* w3pk   = (_Float16*)alloc((size_t)2 * 16 * 32 * 16 * 2);
  _Float16* y1     = (_Float16*)alloc((size_t)NB * PADW * 32 * 2);
  _Float16* y2     = (_Float16*)alloc((size_t)NB * PADW * 32 * 2);

  const int* src = edge;
  const int* dst = edge + N_EDGES;

  // ---- GIN layer 1 (IC=78) ----
  hipMemsetAsync(agg, 0, (size_t)N_NODES * 78 * 4, stream);
  k_gather<<<(N_EDGES * 78 + 255) / 256, 256, 0, stream>>>(x, src, dst, agg, 78);
  k_gin_mlp<<<N_NODES / 128, 128, 0, stream>>>(x, agg, 78, g1_W1, g1_b1, g1_W2, g1_b2,
      bn_g, bn_b, bn_m, bn_v, hA);
  // ---- GIN layers 2..5 ----
  float* hin = hA; float* hout = hB;
  for (int i = 0; i < 4; ++i) {
    hipMemsetAsync(agg, 0, (size_t)N_NODES * 32 * 4, stream);
    k_gather<<<(N_EDGES * 32 + 255) / 256, 256, 0, stream>>>(hin, src, dst, agg, 32);
    k_gin_mlp<<<N_NODES / 128, 128, 0, stream>>>(hin, agg, 32,
        gW1 + (size_t)i * 1024, gb1 + (size_t)i * 32,
        gW2 + (size_t)i * 1024, gb2 + (size_t)i * 32,
        bn_g + (size_t)(i + 1) * 32, bn_b + (size_t)(i + 1) * 32,
        bn_m + (size_t)(i + 1) * 32, bn_v + (size_t)(i + 1) * 32, hout);
    float* tmp = hin; hin = hout; hout = tmp;
  }
  // hin now holds final node features
  hipMemsetAsync(pooled, 0, (size_t)NB * 32 * 4, stream);
  k_pool<<<(N_NODES * 32) / 256, 256, 0, stream>>>(hin, batch, pooled);
  k_dense<<<(NB * 128 + 255) / 256, 256, 0, stream>>>(pooled, fcxd_W, fcxd_b,
      nullptr, nullptr, nullptr, nullptr, xd, NB, 32, 128, 1);

  // ---- pack conv weights to WMMA fragment order ----
  k_pack_w1<<<(2 * 128 * 32 * 16) / 256, 256, 0, stream>>>(c1_W, w1pk);
  k_pack_w23<<<(2 * 16 * 32 * 16) / 256, 256, 0, stream>>>(c2_W, w2pk);
  k_pack_w23<<<(2 * 16 * 32 * 16) / 256, 256, 0, stream>>>(c3_W, w3pk);

  // ---- conv branch (WMMA) ----
  dim3 cg(16, NB);
  k_conv1<<<cg, 128, 0, stream>>>(target, embed, xd, w1pk, c1_b,
      cbn_g, cbn_b, cbn_m, cbn_v, y1);
  k_conv23<<<cg, 128, 0, stream>>>(y1, w2pk, c2_b,
      cbn_g + 32, cbn_b + 32, cbn_m + 32, cbn_v + 32, y2, nullptr, W2OUT, 1);
  hipMemsetAsync(xcm, 0, (size_t)NB * 32 * 4, stream);
  k_conv23<<<cg, 128, 0, stream>>>(y2, w3pk, c3_b,
      cbn_g + 64, cbn_b + 64, cbn_m + 64, cbn_v + 64, nullptr, xcm, W3OUT, 0);

  // ---- head ----
  k_dense<<<(NB * 128 + 255) / 256, 256, 0, stream>>>(xcm, fcxc_W, fcxc_b,
      bnfc_g, bnfc_b, bnfc_m, bnfc_v, xcf, NB, 32, 128, 1);
  k_concat<<<(NB * 256) / 256, 256, 0, stream>>>(xd, xcf, z);
  k_dense<<<(NB * 1024 + 255) / 256, 256, 0, stream>>>(z, fc1_W, fc1_b,
      nullptr, nullptr, nullptr, nullptr, t1, NB, 256, 1024, 1);
  k_dense<<<(NB * 256 + 255) / 256, 256, 0, stream>>>(t1, fc2_W, fc2_b,
      nullptr, nullptr, nullptr, nullptr, t2, NB, 1024, 256, 1);
  k_dense<<<(NB + 255) / 256, 256, 0, stream>>>(t2, out_W, out_b,
      nullptr, nullptr, nullptr, nullptr, (float*)d_out, NB, 256, 1, 0);
}
